// Attention_27994596835457
// MI455X (gfx1250) — matrix operand order repaired and checked by reading, not compiled, not run on hardware
//
#include <hip/hip_runtime.h>
#include <hip/hip_bf16.h>

#define DI __device__ __forceinline__

#if __has_builtin(__builtin_amdgcn_sched_group_barrier)
#define SGB(mask, size, id) __builtin_amdgcn_sched_group_barrier(mask, size, id)
#else
#define SGB(mask, size, id)
#endif

typedef __attribute__((ext_vector_type(16))) __bf16 v16bf;
typedef __attribute__((ext_vector_type(8)))  float  v8f;

union Frag {
  v16bf v;
  uint4 u[2];
};

DI v8f wmma_bf16(v16bf a, v16bf b, v8f c) {
  // D = A(16x32 bf16) * B(32x16 bf16) + C(16x16 f32)
  return __builtin_amdgcn_wmma_f32_16x16x32_bf16(false, a, false, b, (short)0, c,
                                                 false, false);
}

DI uint4 cvt8(float4 a, float4 b) {
  union { __bf16 h[8]; uint4 u; } t;
  t.h[0] = (__bf16)a.x; t.h[1] = (__bf16)a.y; t.h[2] = (__bf16)a.z; t.h[3] = (__bf16)a.w;
  t.h[4] = (__bf16)b.x; t.h[5] = (__bf16)b.y; t.h[6] = (__bf16)b.z; t.h[7] = (__bf16)b.w;
  return t.u;
}

enum { EPI_F32 = 0, EPI_ROPE_BF16 = 1, EPI_TRANS_BF16 = 2 };

// C = A(MxK) * B(NxK)^T, 128x128 block tile, BK=64, 8 waves (2x4), each wave 64x32.
// 2-stage software pipeline: global->regs prefetch of tile kk+64 overlaps the
// 16 WMMAs of tile kk fed from LDS.
template <int MODE, bool A_IS_BF16>
__global__ void __launch_bounds__(256)
gemm_wmma(const void* __restrict__ Ap, const float* __restrict__ B,
          void* __restrict__ Cp, int M, int N, int K,
          const float* __restrict__ cosp, const float* __restrict__ sinp) {
  __shared__ __align__(16) __bf16 As[128][72];
  __shared__ __align__(16) __bf16 Bs[128][72];

  const int tid  = threadIdx.x;
  const int lane = tid & 31;
  const int wave = tid >> 5;
  const int ln   = lane & 15;
  const int lh   = lane >> 4;   // half-wave select
  const int wm   = wave >> 2;   // 0..1 (M direction)
  const int wn   = wave & 3;    // 0..3 (N direction)
  const int mb   = blockIdx.y * 128;
  const int nb   = blockIdx.x * 128;

  // 4 chunks of 8 halves per thread per 128x64 tile
  int crow[4], cpart[4];
#pragma unroll
  for (int i = 0; i < 4; ++i) {
    const int c = tid + i * 256;   // 0..1023
    crow[i]  = c >> 3;             // 0..127
    cpart[i] = (c & 7) * 8;        // 0..56
  }

  float4 fa[4][2], fb[4][2];
  uint4  ua[4];

  auto prefetch = [&](int kk) {
#pragma unroll
    for (int i = 0; i < 4; ++i) {
      if (A_IS_BF16) {
        const __bf16* A = (const __bf16*)Ap;
        ua[i] = *(const uint4*)(A + (size_t)(mb + crow[i]) * K + kk + cpart[i]);
      } else {
        const float*  A = (const float*)Ap;
        const float4* p = (const float4*)(A + (size_t)(mb + crow[i]) * K + kk + cpart[i]);
        fa[i][0] = p[0]; fa[i][1] = p[1];
      }
      const float4* q = (const float4*)(B + (size_t)(nb + crow[i]) * K + kk + cpart[i]);
      fb[i][0] = q[0]; fb[i][1] = q[1];
    }
  };

  v8f acc[4][2];
#pragma unroll
  for (int i = 0; i < 4; ++i)
#pragma unroll
    for (int j = 0; j < 2; ++j)
#pragma unroll
      for (int r = 0; r < 8; ++r) acc[i][j][r] = 0.f;

  prefetch(0);

  for (int kk = 0; kk < K; kk += 64) {
    // stage regs -> LDS
#pragma unroll
    for (int i = 0; i < 4; ++i) {
      *(uint4*)&As[crow[i]][cpart[i]] =
          A_IS_BF16 ? ua[i] : cvt8(fa[i][0], fa[i][1]);
      *(uint4*)&Bs[crow[i]][cpart[i]] = cvt8(fb[i][0], fb[i][1]);
    }
    __syncthreads();
    if (kk + 64 < K) prefetch(kk + 64);   // in flight during WMMAs below

#pragma unroll
    for (int ks = 0; ks < 2; ++ks) {
      Frag af[4], bf[2];
#pragma unroll
      for (int mt = 0; mt < 4; ++mt) {
        const __bf16* p = &As[wm * 64 + mt * 16 + ln][ks * 32];
        af[mt].u[0] = *(const uint4*)(p + lh * 8);        // K 0-7 / 8-15
        af[mt].u[1] = *(const uint4*)(p + 16 + lh * 8);   // K 16-23 / 24-31
      }
#pragma unroll
      for (int nt = 0; nt < 2; ++nt) {
        const __bf16* p = &Bs[wn * 32 + nt * 16 + ln][ks * 32];
        bf[nt].u[0] = *(const uint4*)(p + lh * 8);        // K 0-7 / 8-15
        bf[nt].u[1] = *(const uint4*)(p + 16 + lh * 8);   // K 16-23 / 24-31
      }
#pragma unroll
      for (int mt = 0; mt < 4; ++mt)
#pragma unroll
        for (int nt = 0; nt < 2; ++nt)
          acc[mt][nt] = wmma_bf16(af[mt].v, bf[nt].v, acc[mt][nt]);
      // schedule: batch the 12 LDS fragment reads ahead of the 8 WMMAs
      SGB(0x100, 12, 0);
      SGB(0x008, 8, 0);
    }
    __syncthreads();
  }

#pragma unroll
  for (int mt = 0; mt < 4; ++mt) {
#pragma unroll
    for (int nt = 0; nt < 2; ++nt) {
#pragma unroll
      for (int r = 0; r < 8; ++r) {
        float v = acc[mt][nt][r];
        const int gm = mb + wm * 64 + mt * 16 + r + lh * 8;
        const int gn = nb + wn * 32 + nt * 16 + ln;
        if (MODE == EPI_F32) {
          ((float*)Cp)[(size_t)gm * N + gn] = v;
        } else if (MODE == EPI_ROPE_BF16) {
          const int   hd = gn & 127;          // position within head
          const float c  = cosp[(size_t)gm * 64 + (hd >> 1)];
          const float s  = sinp[(size_t)gm * 64 + (hd >> 1)];
          const float pr = __shfl_xor(v, 1, 32);  // rotation partner (adjacent n)
          const float ov = (hd & 1) ? (pr * s + v * c) : (v * c - pr * s);
          ((__bf16*)Cp)[(size_t)gm * N + gn] = (__bf16)ov;
        } else {  // EPI_TRANS_BF16 : store C^T (N x M)
          ((__bf16*)Cp)[(size_t)gn * M + gm] = (__bf16)v;
        }
      }
    }
  }
}

// Flash attention: grid (S/128, H), 256 threads. Block-uniform causal loop so
// all 8 waves share cooperative LDS staging of K/V tiles (loaded once per
// block, not once per wave) with a 2-stage global->regs->LDS pipeline.
__global__ void __launch_bounds__(256)
attn_fwd(const __bf16* __restrict__ Q, const __bf16* __restrict__ Kb,
         const __bf16* __restrict__ Vt, __bf16* __restrict__ O) {
  __shared__ __align__(16) __bf16 Ks[64][136];   // keys x hd      (17.0 KB)
  __shared__ __align__(16) __bf16 Vs[128][72];   // hd x keys      (18.0 KB)
  __shared__ __align__(16) __bf16 Pl[8][16][72]; // per-wave P     (18.0 KB)

  const int tid  = threadIdx.x;
  const int lane = tid & 31;
  const int wave = tid >> 5;
  const int ln   = lane & 15;
  const int lh   = lane >> 4;
  const int h    = blockIdx.y;
  const int kvh  = h >> 2;                       // H/KVH == 4
  const int qt   = blockIdx.x;
  const int q0   = qt * 128 + wave * 16;

  // cooperative-load chunk coords (4 chunks of 8 halves per thread per tile)
  int krow[4], kpart[4], vrow[4], vpart[4];
#pragma unroll
  for (int i = 0; i < 4; ++i) {
    const int c = tid + i * 256;   // 0..1023
    krow[i] = c >> 4;  kpart[i] = (c & 15) * 8;  // 64 rows x 128 halves
    vrow[i] = c >> 3;  vpart[i] = (c & 7) * 8;   // 128 rows x 64 halves
  }
  uint4 pk[4], pv[4];
  auto prefetchKV = [&](int kb) {
#pragma unroll
    for (int i = 0; i < 4; ++i) {
      pk[i] = *(const uint4*)(Kb + (size_t)(kb + krow[i]) * 1024 + kvh * 128 + kpart[i]);
      pv[i] = *(const uint4*)(Vt + (size_t)(kvh * 128 + vrow[i]) * 2048 + kb + vpart[i]);
    }
  };

  // Q strip as A-fragments (direct from global, layout-matched 16B chunks)
  Frag qf[4];
  const __bf16* qbase = Q + (size_t)(q0 + ln) * 4096 + h * 128;
#pragma unroll
  for (int ks = 0; ks < 4; ++ks) {
    qf[ks].u[0] = *(const uint4*)(qbase + ks * 32 + lh * 8);
    qf[ks].u[1] = *(const uint4*)(qbase + ks * 32 + 16 + lh * 8);
  }

  float m_i[8], l_i[8];
  v8f   o[8];
#pragma unroll
  for (int r = 0; r < 8; ++r) { m_i[r] = -3.0e38f; l_i[r] = 0.f; }
#pragma unroll
  for (int t = 0; t < 8; ++t)
#pragma unroll
    for (int r = 0; r < 8; ++r) o[t][r] = 0.f;

  const float cb = 0.08838834764831845f * 1.4426950408889634f;  // scale*log2(e)
  const int   nk = 2 * qt + 2;   // uniform across the block (causal tile bound)

  prefetchKV(0);

  for (int kt = 0; kt < nk; ++kt) {
    const int kb = kt * 64;
    // stage regs -> LDS (K and V tiles for this key block)
#pragma unroll
    for (int i = 0; i < 4; ++i) {
      *(uint4*)&Ks[krow[i]][kpart[i]] = pk[i];
      *(uint4*)&Vs[vrow[i]][vpart[i]] = pv[i];
    }
    __syncthreads();
    if (kt + 1 < nk) prefetchKV(kb + 64);   // hidden behind the WMMAs below

    // S = Q * K^T  (16 q-rows x 64 keys), K fed from LDS
    float tt[4][8];
#pragma unroll
    for (int nt = 0; nt < 4; ++nt) {
      Frag kf[4];
#pragma unroll
      for (int ks = 0; ks < 4; ++ks) {
        const __bf16* p = &Ks[nt * 16 + ln][ks * 32];
        kf[ks].u[0] = *(const uint4*)(p + lh * 8);
        kf[ks].u[1] = *(const uint4*)(p + 16 + lh * 8);
      }
      v8f sa;
#pragma unroll
      for (int r = 0; r < 8; ++r) sa[r] = 0.f;
#pragma unroll
      for (int ks = 0; ks < 4; ++ks) sa = wmma_bf16(qf[ks].v, kf[ks].v, sa);
      // schedule: batch the 8 LDS fragment reads ahead of the 4 WMMAs
      SGB(0x100, 8, 0);
      SGB(0x008, 4, 0);
      const int kc = kb + nt * 16 + ln;
#pragma unroll
      for (int r = 0; r < 8; ++r) {
        const int qr = q0 + r + lh * 8;
        tt[nt][r] = (kc > qr) ? -3.0e38f : sa[r] * cb;   // causal mask
      }
    }
    // online softmax (rows span the 16 lanes of each half-wave)
    float fs[8];
#pragma unroll
    for (int r = 0; r < 8; ++r) {
      float mx = fmaxf(fmaxf(tt[0][r], tt[1][r]), fmaxf(tt[2][r], tt[3][r]));
#pragma unroll
      for (int off = 8; off > 0; off >>= 1) mx = fmaxf(mx, __shfl_xor(mx, off, 32));
      const float mn = fmaxf(m_i[r], mx);
      fs[r] = exp2f(m_i[r] - mn);
      float ps = 0.f;
#pragma unroll
      for (int nt = 0; nt < 4; ++nt) {
        const float p = exp2f(tt[nt][r] - mn);
        tt[nt][r] = p;
        ps += p;
      }
#pragma unroll
      for (int off = 8; off > 0; off >>= 1) ps += __shfl_xor(ps, off, 32);
      l_i[r] = l_i[r] * fs[r] + ps;
      m_i[r] = mn;
    }
#pragma unroll
    for (int t = 0; t < 8; ++t)
#pragma unroll
      for (int r = 0; r < 8; ++r) o[t][r] *= fs[r];

    // P (C-layout) -> LDS -> A-fragments; per-wave region, wave-local waits only
    asm volatile("s_wait_dscnt 0x0" ::: "memory");
#pragma unroll
    for (int nt = 0; nt < 4; ++nt)
#pragma unroll
      for (int r = 0; r < 8; ++r)
        Pl[wave][r + lh * 8][nt * 16 + ln] = (__bf16)tt[nt][r];
    asm volatile("s_wait_dscnt 0x0" ::: "memory");
    Frag pf[2];
#pragma unroll
    for (int ks = 0; ks < 2; ++ks) {
      const __bf16* p = &Pl[wave][ln][ks * 32];
      pf[ks].u[0] = *(const uint4*)(p + lh * 8);
      pf[ks].u[1] = *(const uint4*)(p + 16 + lh * 8);
    }
    // O += P * V, V fed from LDS (stored hd-major -> contiguous key dim)
#pragma unroll
    for (int t2 = 0; t2 < 8; ++t2) {
      Frag vf[2];
#pragma unroll
      for (int ks = 0; ks < 2; ++ks) {
        const __bf16* p = &Vs[t2 * 16 + ln][ks * 32];
        vf[ks].u[0] = *(const uint4*)(p + lh * 8);
        vf[ks].u[1] = *(const uint4*)(p + 16 + lh * 8);
      }
      o[t2] = wmma_bf16(pf[0].v, vf[0].v, o[t2]);
      o[t2] = wmma_bf16(pf[1].v, vf[1].v, o[t2]);
      // schedule: batch the 4 LDS fragment reads ahead of the 2 WMMAs
      SGB(0x100, 4, 0);
      SGB(0x008, 2, 0);
    }
    __syncthreads();   // protect Ks/Vs for next iteration's staging
  }

#pragma unroll
  for (int t2 = 0; t2 < 8; ++t2) {
#pragma unroll
    for (int r = 0; r < 8; ++r) {
      const int s   = q0 + r + lh * 8;
      const int col = h * 128 + t2 * 16 + ln;
      O[(size_t)s * 4096 + col] = (__bf16)(o[t2][r] / l_i[r]);
    }
  }
}

extern "C" void kernel_launch(void* const* d_in, const int* in_sizes, int n_in,
                              void* d_out, int out_size, void* d_ws, size_t ws_size,
                              hipStream_t stream) {
  (void)in_sizes; (void)n_in; (void)out_size; (void)ws_size;
  const float* x  = (const float*)d_in[0];
  const float* fc = (const float*)d_in[1];
  const float* fsn= (const float*)d_in[2];
  const float* wq = (const float*)d_in[3];
  const float* wk = (const float*)d_in[4];
  const float* wv = (const float*)d_in[5];
  const float* wo = (const float*)d_in[6];
  float* out = (float*)d_out;

  char* ws = (char*)d_ws;
  __bf16* Qb  = (__bf16*)(ws);                         // 2048x4096 bf16 (16 MB)
  __bf16* Kb  = (__bf16*)(ws + (size_t)16777216);      // 2048x1024 bf16 (4 MB)
  __bf16* Vtb = (__bf16*)(ws + (size_t)20971520);      // 1024x2048 bf16 (4 MB, V^T)
  __bf16* Ab  = (__bf16*)(ws + (size_t)25165824);      // 2048x4096 bf16 (16 MB)

  dim3 blk(256);
  // Q = rope(x @ wq^T)
  gemm_wmma<EPI_ROPE_BF16, false><<<dim3(32, 16), blk, 0, stream>>>(
      x, wq, Qb, 2048, 4096, 4096, fc, fsn);
  // K = rope(x @ wk^T)
  gemm_wmma<EPI_ROPE_BF16, false><<<dim3(8, 16), blk, 0, stream>>>(
      x, wk, Kb, 2048, 1024, 4096, fc, fsn);
  // V^T = (x @ wv^T)^T
  gemm_wmma<EPI_TRANS_BF16, false><<<dim3(8, 16), blk, 0, stream>>>(
      x, wv, Vtb, 2048, 1024, 4096, nullptr, nullptr);
  // attention
  attn_fwd<<<dim3(16, 32), blk, 0, stream>>>(Qb, Kb, Vtb, Ab);
  // out = attn @ wo^T (fp32)
  gemm_wmma<EPI_F32, true><<<dim3(32, 16), blk, 0, stream>>>(
      Ab, wo, out, 2048, 4096, 4096, nullptr, nullptr);
}
